// MultiHeadAttention_33629593927747
// MI455X (gfx1250) — matrix operand order repaired and checked by reading, not compiled, not run on hardware
//
#include <hip/hip_runtime.h>
#include <hip/hip_bf16.h>

#define SLEN   4096
#define HEADS  12
#define DMODEL 768
#define DKH    64

typedef __attribute__((ext_vector_type(16))) __bf16 v16bf;
typedef __attribute__((ext_vector_type(8)))  float  v8f;

union AFrag { v16bf v; __bf16 e[16]; unsigned int u[8]; };
union CFrag { v8f   v; float f[8]; };
union Pack8 { __bf16 e[8]; uint4 q; };

#if __has_builtin(__builtin_amdgcn_global_load_async_to_lds_b128)
#define HAVE_ASYNC 1
typedef int v4ia __attribute__((ext_vector_type(4)));
typedef __attribute__((address_space(1))) v4ia gas_v4i;   // global int4
typedef __attribute__((address_space(3))) v4ia las_v4i;   // LDS int4
#else
#define HAVE_ASYNC 0
#endif

__device__ __forceinline__ void wait_async0() {
#if HAVE_ASYNC
#if __has_builtin(__builtin_amdgcn_s_wait_asynccnt)
  __builtin_amdgcn_s_wait_asynccnt(0);
#else
  asm volatile("s_wait_asynccnt 0x0" ::: "memory");
#endif
#endif
}

__device__ __forceinline__ void ld128(unsigned int* d, const void* p) {
  uint4 t = *reinterpret_cast<const uint4*>(p);
  d[0] = t.x; d[1] = t.y; d[2] = t.z; d[3] = t.w;
}

// ---------------------------------------------------------------------------
// Kernel 0: convert the three weight matrices f32 -> bf16 (one pass).
// ---------------------------------------------------------------------------
__global__ void __launch_bounds__(256) cvt_w_kernel(
    const float* __restrict__ Wq, const float* __restrict__ Wk,
    const float* __restrict__ Wv,
    __bf16* __restrict__ wbq, __bf16* __restrict__ wbk, __bf16* __restrict__ wbv)
{
  const float* src;
  __bf16* dst;
  if (blockIdx.y == 0)      { src = Wq; dst = wbq; }
  else if (blockIdx.y == 1) { src = Wk; dst = wbk; }
  else                      { src = Wv; dst = wbv; }

  const size_t i = ((size_t)blockIdx.x * 256 + threadIdx.x) * 8;
  float4 a = *reinterpret_cast<const float4*>(src + i);
  float4 b = *reinterpret_cast<const float4*>(src + i + 4);
  Pack8 p;
  p.e[0] = (__bf16)a.x; p.e[1] = (__bf16)a.y;
  p.e[2] = (__bf16)a.z; p.e[3] = (__bf16)a.w;
  p.e[4] = (__bf16)b.x; p.e[5] = (__bf16)b.y;
  p.e[6] = (__bf16)b.z; p.e[7] = (__bf16)b.w;
  *reinterpret_cast<uint4*>(dst + i) = p.q;
}

// ---------------------------------------------------------------------------
// Kernel 1: QKV projection.  out = X*W + b, bf16 WMMA, f32 accumulate.
// Wave tile: 16 rows x 64 cols.  grid = (8192/128, 768/64, 3); z selects Q/K/V.
// Q,V written bf16 [B,H,S,DK]; K written bf16 [B,H,DK,S] (transposed).
// ---------------------------------------------------------------------------
__global__ void __launch_bounds__(256) qkv_proj_kernel(
    const float* __restrict__ xq, const float* __restrict__ xk, const float* __restrict__ xv,
    const __bf16* __restrict__ Wbq, const float* __restrict__ bq,
    const __bf16* __restrict__ Wbk, const float* __restrict__ bk,
    const __bf16* __restrict__ Wbv, const float* __restrict__ bv,
    __bf16* __restrict__ wsq, __bf16* __restrict__ wsk, __bf16* __restrict__ wsv)
{
  const int lane  = threadIdx.x & 31;
  const int wv_id = threadIdx.x >> 5;
  const int hi    = lane >> 4;
  const int lm    = lane & 15;

  const float* X; const __bf16* W; const float* bias; __bf16* out; int kmode;
  if (blockIdx.z == 0)      { X = xq; W = Wbq; bias = bq; out = wsq; kmode = 0; }
  else if (blockIdx.z == 1) { X = xk; W = Wbk; bias = bk; out = wsk; kmode = 1; }
  else                      { X = xv; W = Wbv; bias = bv; out = wsv; kmode = 0; }

  const int rowBase = blockIdx.x * 128 + wv_id * 16;   // [0, 8192)
  const int colBase = blockIdx.y * 64;                 // [0, 768)

  CFrag acc[4];
  #pragma unroll
  for (int n = 0; n < 4; ++n)
    #pragma unroll
    for (int r = 0; r < 8; ++r) acc[n].f[r] = 0.0f;

  for (int k0 = 0; k0 < DMODEL; k0 += 32) {
    // A fragment: X rows rowBase+lm; lo lanes K={0..7,16..23}, hi K={8..15,24..31}
    AFrag a;
    const float* xr = X + (size_t)(rowBase + lm) * DMODEL + k0 + hi * 8;
    float4 x0 = *reinterpret_cast<const float4*>(xr);
    float4 x1 = *reinterpret_cast<const float4*>(xr + 4);
    float4 x2 = *reinterpret_cast<const float4*>(xr + 16);
    float4 x3 = *reinterpret_cast<const float4*>(xr + 20);
    float xvv[16] = {x0.x,x0.y,x0.z,x0.w, x1.x,x1.y,x1.z,x1.w,
                     x2.x,x2.y,x2.z,x2.w, x3.x,x3.y,x3.z,x3.w};
    #pragma unroll
    for (int j = 0; j < 16; ++j) a.e[j] = (__bf16)xvv[j];

    // B fragments: lane L holds W column colBase+n*16+lm, K={hi*8+0..7, 16+hi*8+0..7}
    const __bf16* wr = W + (size_t)(k0 + hi * 8) * DMODEL + colBase + lm;
    #pragma unroll
    for (int n = 0; n < 4; ++n) {
      AFrag bf;
      #pragma unroll
      for (int j = 0; j < 16; ++j)
        bf.e[j] = wr[(size_t)((j & 7) + (j >> 3) * 16) * DMODEL + n * 16];
      acc[n].v = __builtin_amdgcn_wmma_f32_16x16x32_bf16(
          false, a.v, false, bf.v, (short)0, acc[n].v, false, false);
    }
  }

  // Epilogue: bias add, convert to bf16, store per-head layout.
  #pragma unroll
  for (int n = 0; n < 4; ++n) {
    const int col = colBase + n * 16 + lm;
    const float bi = bias[col];
    const int hh = col >> 6;
    const int dk = col & 63;
    #pragma unroll
    for (int r = 0; r < 8; ++r) {
      const int row = rowBase + r + 8 * hi;
      const int bb  = row >> 12;
      const int ss  = row & 4095;
      const float val = acc[n].f[r] + bi;
      size_t idx;
      if (kmode) idx = (((size_t)(bb * HEADS + hh) * DKH + dk) * SLEN + ss); // K: [B,H,DK,S]
      else       idx = (((size_t)(bb * HEADS + hh) * SLEN + ss) * DKH + dk); // Q,V: [B,H,S,DK]
      out[idx] = (__bf16)val;
    }
  }
}

// ---------------------------------------------------------------------------
// Stage one 64-key K/V tile into LDS (16 KB), whole block cooperating.
// K tile layout [feat(64)][key(64)], V tile layout [key(64)][feat(64)].
// Async path: GLOBAL_LOAD_ASYNC_TO_LDS_B128 (ASYNCcnt); fallback: reg-staged.
// ---------------------------------------------------------------------------
__device__ __forceinline__ void stage_tile(const __bf16* gK, const __bf16* gV,
                                           __bf16* lK, __bf16* lV, int tid)
{
  #pragma unroll
  for (int j = 0; j < 2; ++j) {
    const int ch  = tid + j * 256;       // 512 chunks of 16B per tile
    const int row = ch >> 3;
    const int off = (ch & 7) * 8;
#if HAVE_ASYNC
    __builtin_amdgcn_global_load_async_to_lds_b128(
        (gas_v4i*)(gK + (size_t)row * SLEN + off),
        (las_v4i*)(lK + ch * 8), 0, 0);
    __builtin_amdgcn_global_load_async_to_lds_b128(
        (gas_v4i*)(gV + row * DKH + off),
        (las_v4i*)(lV + ch * 8), 0, 0);
#else
    *reinterpret_cast<uint4*>(lK + ch * 8) =
        *reinterpret_cast<const uint4*>(gK + (size_t)row * SLEN + off);
    *reinterpret_cast<uint4*>(lV + ch * 8) =
        *reinterpret_cast<const uint4*>(gV + row * DKH + off);
#endif
  }
}

// ---------------------------------------------------------------------------
// Kernel 2: flash attention.  One wave owns 16 query rows; block = 8 waves,
// all sharing (b,h).  K/V tiles staged in double-buffered LDS via async DMA.
// grid = (4096/128, H, B).  Per 64-key tile: 8 WMMA (QK^T) + 8 WMMA (PV).
// ---------------------------------------------------------------------------
__global__ void __launch_bounds__(256) flash_attn_kernel(
    const __bf16* __restrict__ Qw,   // [B,H,S,DK]  bf16
    const __bf16* __restrict__ Kt,   // [B,H,DK,S]  bf16 (transposed)
    const __bf16* __restrict__ Vw,   // [B,H,S,DK]  bf16
    const float* __restrict__ mask,  // [B,S]
    float* __restrict__ out)         // [B,S,D] f32
{
  __shared__ __bf16 kLds[2][64 * 64];   // [buf][feat][key]   8KB each
  __shared__ __bf16 vLds[2][64 * 64];   // [buf][key][feat]   8KB each
  __shared__ __bf16 pLds[8][16 * 64];   // per-wave P tile    2KB each

  const int tid   = threadIdx.x;
  const int lane  = tid & 31;
  const int wv_id = tid >> 5;
  const int hi    = lane >> 4;
  const int lm    = lane & 15;
  const int b     = blockIdx.z;
  const int h     = blockIdx.y;
  const int qBase = blockIdx.x * 128 + wv_id * 16;

  const size_t bh = (size_t)(b * HEADS + h);
  const __bf16* Qp = Qw + bh * SLEN * DKH;
  const __bf16* Kp = Kt + bh * DKH * SLEN;
  const __bf16* Vp = Vw + bh * SLEN * DKH;
  const float* mp = mask + (size_t)b * SLEN;

  // Q A-fragments: 2 chunks of K=32 over DK=64 (per-half K pattern).
  AFrag qa[2];
  {
    const __bf16* qr = Qp + (size_t)(qBase + lm) * DKH;
    #pragma unroll
    for (int c = 0; c < 2; ++c) {
      const int c0 = c * 32 + hi * 8;
      ld128(qa[c].u,     qr + c0);
      ld128(qa[c].u + 4, qr + c0 + 16);
    }
  }

  float mrow[8], lrow[8];
  CFrag o[4];
  #pragma unroll
  for (int r = 0; r < 8; ++r) { mrow[r] = -3.0e38f; lrow[r] = 0.0f; }
  #pragma unroll
  for (int n = 0; n < 4; ++n)
    #pragma unroll
    for (int r = 0; r < 8; ++r) o[n].f[r] = 0.0f;

  __bf16* pl = pLds[wv_id];

  // Prefetch tile 0.
  stage_tile(Kp, Vp, kLds[0], vLds[0], tid);

  for (int it = 0; it < SLEN / 64; ++it) {
    const int kb = it * 64;
    wait_async0();        // tile `it` resident (each wave waits its own copies)
    __syncthreads();      // ..and everyone else's; prev compute finished

    if (it + 1 < SLEN / 64) {
      const int kn = kb + 64;
      stage_tile(Kp + kn, Vp + (size_t)kn * DKH,
                 kLds[(it + 1) & 1], vLds[(it + 1) & 1], tid);
    }

    const __bf16* lK = kLds[it & 1];
    const __bf16* lV = vLds[it & 1];

    // ---- scores = Q * K^T for 64 keys (4 N-tiles x 2 K-chunks) ----
    CFrag s[4];
    #pragma unroll
    for (int n = 0; n < 4; ++n) {
      #pragma unroll
      for (int r = 0; r < 8; ++r) s[n].f[r] = 0.0f;
      #pragma unroll
      for (int c = 0; c < 2; ++c) {
        AFrag kf;  // B frag: lane = key n*16+lm, features c*32+{hi*8+0..7, 16+hi*8+0..7}
        const __bf16* kr = lK + (c * 32 + hi * 8) * 64 + n * 16 + lm;
        #pragma unroll
        for (int j = 0; j < 16; ++j) kf.e[j] = kr[((j & 7) + (j >> 3) * 16) * 64];
        s[n].v = __builtin_amdgcn_wmma_f32_16x16x32_bf16(
            false, qa[c].v, false, kf.v, (short)0, s[n].v, false, false);
      }
    }

    // ---- scale + mask adder (column = key = lane%16 within tile) ----
    #pragma unroll
    for (int n = 0; n < 4; ++n) {
      const float add = (1.0f - mp[kb + n * 16 + lm]) * -10000.0f;
      #pragma unroll
      for (int r = 0; r < 8; ++r) s[n].f[r] = s[n].f[r] * 0.125f + add;
    }

    // ---- online softmax (row stats live in C-layout row mapping) ----
    float tm[8];
    #pragma unroll
    for (int r = 0; r < 8; ++r)
      tm[r] = fmaxf(fmaxf(s[0].f[r], s[1].f[r]), fmaxf(s[2].f[r], s[3].f[r]));
    #pragma unroll
    for (int off = 1; off < 16; off <<= 1)
      #pragma unroll
      for (int r = 0; r < 8; ++r) tm[r] = fmaxf(tm[r], __shfl_xor(tm[r], off, 32));

    float alpha[8];
    #pragma unroll
    for (int r = 0; r < 8; ++r) {
      const float mn = fmaxf(mrow[r], tm[r]);
      alpha[r] = __expf(mrow[r] - mn);
      mrow[r] = mn;
    }

    float ts[8];
    #pragma unroll
    for (int r = 0; r < 8; ++r) ts[r] = 0.0f;
    #pragma unroll
    for (int n = 0; n < 4; ++n)
      #pragma unroll
      for (int r = 0; r < 8; ++r) {
        const float p = __expf(s[n].f[r] - mrow[r]);
        s[n].f[r] = p;
        ts[r] += p;
      }
    #pragma unroll
    for (int off = 1; off < 16; off <<= 1)
      #pragma unroll
      for (int r = 0; r < 8; ++r) ts[r] += __shfl_xor(ts[r], off, 32);
    #pragma unroll
    for (int r = 0; r < 8; ++r) lrow[r] = lrow[r] * alpha[r] + ts[r];
    #pragma unroll
    for (int n = 0; n < 4; ++n)
      #pragma unroll
      for (int r = 0; r < 8; ++r) o[n].f[r] *= alpha[r];

    // ---- P: C-layout -> LDS (row-major 16x64 bf16) -> A-layout ----
    #pragma unroll
    for (int n = 0; n < 4; ++n)
      #pragma unroll
      for (int r = 0; r < 8; ++r)
        pl[(r + 8 * hi) * 64 + n * 16 + lm] = (__bf16)s[n].f[r];
    asm volatile("s_wait_dscnt 0x0" ::: "memory");   // intra-wave LDS RAW

    AFrag pa[2];
    #pragma unroll
    for (int c = 0; c < 2; ++c) {
      const __bf16* pr = pl + lm * 64 + c * 32 + hi * 8;
      ld128(pa[c].u,     pr);
      ld128(pa[c].u + 4, pr + 16);
    }

    // ---- O += P * V (4 dk-tiles x 2 key-chunks) ----
    #pragma unroll
    for (int n = 0; n < 4; ++n)
      #pragma unroll
      for (int c = 0; c < 2; ++c) {
        AFrag vf;  // B frag: lane = dk n*16+lm, keys c*32+{hi*8+0..7, 16+hi*8+0..7}
        const __bf16* vr = lV + (c * 32 + hi * 8) * 64 + n * 16 + lm;
        #pragma unroll
        for (int j = 0; j < 16; ++j) vf.e[j] = vr[((j & 7) + (j >> 3) * 16) * 64];
        o[n].v = __builtin_amdgcn_wmma_f32_16x16x32_bf16(
            false, pa[c].v, false, vf.v, (short)0, o[n].v, false, false);
      }
  }

  // ---- normalize and store: out[b, s, h*64 + dk] ----
  float inv[8];
  #pragma unroll
  for (int r = 0; r < 8; ++r) inv[r] = 1.0f / lrow[r];
  #pragma unroll
  for (int n = 0; n < 4; ++n)
    #pragma unroll
    for (int r = 0; r < 8; ++r) {
      const int row = qBase + r + 8 * hi;
      out[((size_t)b * SLEN + row) * DMODEL + h * 64 + n * 16 + lm] =
          o[n].f[r] * inv[r];
    }
}

extern "C" void kernel_launch(void* const* d_in, const int* in_sizes, int n_in,
                              void* d_out, int out_size, void* d_ws, size_t ws_size,
                              hipStream_t stream) {
  (void)in_sizes; (void)n_in; (void)out_size; (void)ws_size;
  const float* q  = (const float*)d_in[0];
  const float* k  = (const float*)d_in[1];
  const float* v  = (const float*)d_in[2];
  const float* mk = (const float*)d_in[3];
  const float* Wq = (const float*)d_in[4];
  const float* bq = (const float*)d_in[5];
  const float* Wk = (const float*)d_in[6];
  const float* bk = (const float*)d_in[7];
  const float* Wv = (const float*)d_in[8];
  const float* bv = (const float*)d_in[9];
  float* out = (float*)d_out;

  const size_t elems = (size_t)2 * SLEN * DMODEL;      // 6,291,456 per tensor
  const size_t welem = (size_t)DMODEL * DMODEL;        // 589,824 per weight
  __bf16* wsq = (__bf16*)d_ws;                         // Q  [B,H,S,DK]
  __bf16* wsk = wsq + elems;                           // Kt [B,H,DK,S]
  __bf16* wsv = wsk + elems;                           // V  [B,H,S,DK]
  __bf16* wbq = wsv + elems;                           // Wq bf16 [D,D]
  __bf16* wbk = wbq + welem;                           // Wk bf16
  __bf16* wbv = wbk + welem;                           // Wv bf16

  dim3 gc(welem / (256 * 8), 3);   // 288 x 3
  cvt_w_kernel<<<gc, 256, 0, stream>>>(Wq, Wk, Wv, wbq, wbk, wbv);

  dim3 gp(64, 12, 3);     // 8192/128 row-blocks, 768/64 col-blocks, {Q,K,V}
  qkv_proj_kernel<<<gp, 256, 0, stream>>>(q, k, v, wbq, bq, wbk, bk, wbv, bv,
                                          wsq, wsk, wsv);

  dim3 ga(32, 12, 2);     // 4096/128 q-blocks, H, B
  flash_attn_kernel<<<ga, 256, 0, stream>>>(wsq, wsk, wsv, mk, out);
}
